// amp_GCN_Geo_79096117723169
// MI455X (gfx1250) — matrix operand order repaired and checked by reading, not compiled, not run on hardware
//
#include <hip/hip_runtime.h>

typedef __attribute__((ext_vector_type(16))) _Float16 v16h;
typedef __attribute__((ext_vector_type(8)))  float    v8f;

static constexpr int Ecnt = 100000;
static constexpr int Ncnt = 50000;

// A-fragment element permutation: within a 32-wide K chunk, lane-half 0 reads
// k = {0..7,16..23}, half 1 reads k = {8..15,24..31}. Storing rows with bits
// 3<->4 swapped makes each half's 16 elements one contiguous 32B v16h.
__device__ __forceinline__ int swap34(int c) {
  return (c & ~24) | ((c & 8) << 1) | ((c & 16) >> 1);
}

// ---------------------------------------------------------------------------
// Generic zero fill
// ---------------------------------------------------------------------------
__global__ void zero_f32(float* __restrict__ p, int n) {
  int i = blockIdx.x * blockDim.x + threadIdx.x;
  if (i < n) p[i] = 0.f;
}

// ---------------------------------------------------------------------------
// NNConv edge GEMM via WMMA:
//   msg[e, tile_base*32 + 0..31] = Sum_k z[e,k] * Wk[k, col],
//   z[e, s*32+f] = ea[e,s] * hin[src[e], f]   (s<16),   s==16 -> bias row (ea=1)
// A fragment = (staged f16 row, constant over K) * ea scalar -> v_pk_mul_f16.
// Scatter-add into agg[dst[e]]. 128 edges/block (8 waves x 16 edges).
// ---------------------------------------------------------------------------
__global__ void __launch_bounds__(256)
nnconv_edge_wmma(const float* __restrict__ hin,   // [N, 32]
                 const float* __restrict__ eattr, // [E, 16]
                 const int*   __restrict__ eidx,  // [2, E]
                 const float* __restrict__ We,    // [16, 32*fout]
                 const float* __restrict__ be,    // [32*fout]
                 float* __restrict__ agg,         // [N, fout]
                 int fout, int tile_base)
{
  __shared__ __align__(32) _Float16 sWb[17 * 2 * 32 * 16]; // [s][t][lane][j]
  __shared__ __align__(32) _Float16 sXh[8][16][32];        // permuted f16 rows
  __shared__ _Float16 sEAh[8][16][16];

  const int tid = threadIdx.x;
  const int w   = tid >> 5;
  const int ln  = tid & 31;
  const int m   = ln & 15;
  const int hf  = ln >> 4;

  // Stage B fragments (block-shared): B layout k = half*8 + (j&7) + (j&8)*2, col = lane&15
  for (int idx = tid; idx < 17 * 2 * 32 * 16; idx += 256) {
    int j  = idx & 15;
    int r  = idx >> 4;
    int ls = r & 31; r >>= 5;
    int t  = r & 1;
    int s  = r >> 1;
    int half = ls >> 4, nl = ls & 15;
    int row = half * 8 + (j & 7) + (j & 8) * 2;   // fin index 0..31
    int col = tile_base * 32 + t * 16 + nl;       // output column
    float v = (s < 16) ? We[s * (32 * fout) + row * fout + col]
                       : be[row * fout + col];
    sWb[idx] = (_Float16)v;
  }

  // Stage per-wave edge rows (gather, converted + permuted) and attributes
  const int base = blockIdx.x * 128 + w * 16;
  for (int i = ln; i < 16 * 32; i += 32) {
    int r = i >> 5, c = i & 31;
    int e = base + r;
    float v = 0.f;
    if (e < Ecnt) { int s0 = eidx[e]; v = hin[(size_t)s0 * 32 + c]; }
    sXh[w][r][swap34(c)] = (_Float16)v;
  }
  for (int i = ln; i < 16 * 16; i += 32) {
    int r = i >> 4, c = i & 15;
    int e = base + r;
    sEAh[w][r][c] = (_Float16)((e < Ecnt) ? eattr[(size_t)e * 16 + c] : 0.f);
  }
  __syncthreads();

  // Base A fragment: constant across all K steps (one 32B LDS load)
  const v16h ax = *(const v16h*)&sXh[w][m][hf * 16];

  v8f acc0 = {}; v8f acc1 = {};
  for (int s = 0; s < 17; ++s) {
    _Float16 eash = (s < 16) ? sEAh[w][m][s] : (_Float16)1.0f;
    v16h a = ax * eash;                           // 8x v_pk_mul_f16
    v16h b0 = *(const v16h*)&sWb[((s * 2 + 0) * 32 + ln) * 16];
    v16h b1 = *(const v16h*)&sWb[((s * 2 + 1) * 32 + ln) * 16];
    acc0 = __builtin_amdgcn_wmma_f32_16x16x32_f16(false, a, false, b0,
                                                  (short)0, acc0, false, false);
    acc1 = __builtin_amdgcn_wmma_f32_16x16x32_f16(false, a, false, b1,
                                                  (short)0, acc1, false, false);
  }

#pragma unroll
  for (int r = 0; r < 8; ++r) {
    int row = hf * 8 + r;
    int e = base + row;
    if (e < Ecnt) {
      int d = eidx[Ecnt + e];
      atomicAdd(&agg[(size_t)d * fout + tile_base * 32 + m],      acc0[r]);
      atomicAdd(&agg[(size_t)d * fout + tile_base * 32 + 16 + m], acc1[r]);
    }
  }
}

// ---------------------------------------------------------------------------
// hout = relu(agg + hin @ root + bias)
// ---------------------------------------------------------------------------
__global__ void node_update(const float* __restrict__ hin,  // [N, fin]
                            const float* __restrict__ agg,  // [N, fout]
                            const float* __restrict__ root, // [fin, fout]
                            const float* __restrict__ bias, // [fout]
                            float* __restrict__ hout,       // [N, fout]
                            int n, int fin, int fout)
{
  int node = blockIdx.x * blockDim.x + threadIdx.x;
  if (node >= n) return;
  float xr[32];
  for (int f = 0; f < fin; ++f) xr[f] = hin[(size_t)node * fin + f];
  for (int o = 0; o < fout; ++o) {
    float s = bias[o] + agg[(size_t)node * fout + o];
    for (int f = 0; f < fin; ++f) s += xr[f] * root[f * fout + o]; // uniform -> SMEM
    hout[(size_t)node * fout + o] = fmaxf(s, 0.f);
  }
}

// ---------------------------------------------------------------------------
// Per-graph attention over atoms + segment-sum by monomer label, then
// concat with aa_features into h_aa [B,50,160] (col 159 = zero pad).
// ---------------------------------------------------------------------------
__global__ void __launch_bounds__(256)
atom_readout(const float* __restrict__ h2,      // [N, 64]
             const float* __restrict__ aaf,     // [B, 50, 95]
             const int*   __restrict__ labels,  // [B, 1000]
             const float* __restrict__ Wa, const float* __restrict__ ba,
             float* __restrict__ h_aa)          // [B, 50, 160]
{
  __shared__ float sScore[1000];
  __shared__ float sRed[256];
  __shared__ float sAA[50 * 64];
  __shared__ float sWa[64];
  int b = blockIdx.x, tid = threadIdx.x;
  if (tid < 64) sWa[tid] = Wa[tid];
  __syncthreads();
  for (int i = tid; i < 1000; i += 256) {
    const float* hr = h2 + (size_t)(b * 1000 + i) * 64;
    float s = ba[0];
    for (int o = 0; o < 64; ++o) s += hr[o] * sWa[o];
    sScore[i] = s;
  }
  __syncthreads();
  float loc = -3.0e38f;
  for (int i = tid; i < 1000; i += 256) loc = fmaxf(loc, sScore[i]);
  sRed[tid] = loc; __syncthreads();
  for (int st = 128; st > 0; st >>= 1) {
    if (tid < st) sRed[tid] = fmaxf(sRed[tid], sRed[tid + st]);
    __syncthreads();
  }
  float mx = sRed[0];
  __syncthreads();
  float ls = 0.f;
  for (int i = tid; i < 1000; i += 256) {
    float e = __expf(sScore[i] - mx); sScore[i] = e; ls += e;
  }
  sRed[tid] = ls; __syncthreads();
  for (int st = 128; st > 0; st >>= 1) {
    if (tid < st) sRed[tid] += sRed[tid + st];
    __syncthreads();
  }
  float inv = 1.f / sRed[0];
  for (int i = tid; i < 50 * 64; i += 256) sAA[i] = 0.f;
  __syncthreads();
  for (int idx = tid; idx < 1000 * 64; idx += 256) {
    int i = idx >> 6, o = idx & 63;
    int lbl = labels[b * 1000 + i];
    atomicAdd(&sAA[lbl * 64 + o],
              sScore[i] * inv * h2[(size_t)(b * 1000 + i) * 64 + o]);
  }
  __syncthreads();
  for (int idx = tid; idx < 50 * 160; idx += 256) {
    int mm = idx / 160, c = idx - mm * 160;
    float v;
    if (c < 64)       v = sAA[mm * 64 + c];
    else if (c < 159) v = aaf[(size_t)(b * 50 + mm) * 95 + (c - 64)];
    else              v = 0.f;
    h_aa[(size_t)(b * 50 + mm) * 160 + c] = v;
  }
}

// ---------------------------------------------------------------------------
// Rearrange [arma_init_w | arma_root_w] into f16 B-fragment table:
//   wb[s][ct][lane][j] = Wcat[k = s*32 + (lane>>4)*8 + (j&7) + (j&8)*2][n = ct*16 + (lane&15)]
// Wcat cols: n<384 -> init (k3*128+o);  n>=384 -> root_t (384 + t*384 + k3*128 + o)
// ---------------------------------------------------------------------------
__global__ void prep_wbarma(const float* __restrict__ initw, // [3,159,128]
                            const float* __restrict__ rootw, // [6,3,159,128]
                            _Float16* __restrict__ wb)       // [5][168][32][16]
{
  int idx = blockIdx.x * blockDim.x + threadIdx.x;
  if (idx >= 5 * 168 * 32 * 16) return;
  int j = idx & 15;
  int r = idx >> 4;
  int ln = r & 31; r >>= 5;
  int ct = r % 168; int s = r / 168;
  int hf = ln >> 4, nl = ln & 15;
  int k = s * 32 + hf * 8 + (j & 7) + (j & 8) * 2;
  int n = ct * 16 + nl;
  float v = 0.f;
  if (k < 159) {
    if (n < 384) {
      int k3 = n >> 7, o = n & 127;
      v = initw[((size_t)k3 * 159 + k) * 128 + o];
    } else {
      int n2 = n - 384; int t = n2 / 384; int rem = n2 - t * 384;
      int k3 = rem >> 7, o = rem & 127;
      v = rootw[((size_t)(t * 3 + k3) * 159 + k) * 128 + o];
    }
  }
  wb[idx] = (_Float16)v;
}

// ---------------------------------------------------------------------------
// WMMA GEMM: outc[2500, 2688] = h_aa[2500, 160] x Wcat[160, 2688]
// Block = 8 waves, 16 rows x 256 cols per block. A tile staged in LDS as
// f16, pre-permuted -> per-step A fragment is a single 32B LDS load.
// ---------------------------------------------------------------------------
__global__ void __launch_bounds__(256)
arma_gemm_wmma(const float* __restrict__ h_aa,    // [2500, 160]
               const _Float16* __restrict__ wb,   // [5][168][32][16]
               float* __restrict__ outc)          // [2500, 2688]
{
  __shared__ __align__(32) _Float16 sAh[16][160];  // permuted per 32-chunk
  int tid = threadIdx.x;
  int rowBase = blockIdx.x * 16;
  for (int i = tid; i < 16 * 160; i += 256) {
    int rr = i / 160, c = i - rr * 160;
    int row = rowBase + rr;
    float v = (row < 2500) ? h_aa[(size_t)row * 160 + c] : 0.f;
    int chunk = c >> 5, off = c & 31;
    sAh[rr][chunk * 32 + swap34(off)] = (_Float16)v;
  }
  __syncthreads();
  int w = tid >> 5, ln = tid & 31;
  int m = ln & 15, hf = ln >> 4;
  int ct0 = blockIdx.y * 16 + w * 2;
  int ct1 = ct0 + 1;
  bool ok0 = ct0 < 168, ok1 = ct1 < 168;   // wave-uniform guards (EXEC stays full)
  v8f acc0 = {}; v8f acc1 = {};
  for (int s = 0; s < 5; ++s) {
    v16h a = *(const v16h*)&sAh[m][s * 32 + hf * 16];
    if (ok0) {
      v16h b0 = *(const v16h*)&wb[((size_t)(s * 168 + ct0) * 32 + ln) * 16];
      acc0 = __builtin_amdgcn_wmma_f32_16x16x32_f16(false, a, false, b0,
                                                    (short)0, acc0, false, false);
    }
    if (ok1) {
      v16h b1 = *(const v16h*)&wb[((size_t)(s * 168 + ct1) * 32 + ln) * 16];
      acc1 = __builtin_amdgcn_wmma_f32_16x16x32_f16(false, a, false, b1,
                                                    (short)0, acc1, false, false);
    }
  }
#pragma unroll
  for (int r = 0; r < 8; ++r) {
    int row = rowBase + hf * 8 + r;
    if (row < 2500) {
      if (ok0) outc[(size_t)row * 2688 + ct0 * 16 + m] = acc0[r];
      if (ok1) outc[(size_t)row * 2688 + ct1 * 16 + m] = acc1[r];
    }
  }
}

// ---------------------------------------------------------------------------
// ARMA recurrence: one block per graph, out/scratch resident in LDS.
// rootc holds precomputed x0@init (cols k*128+o) and x0@root_t
// (cols 384 + t*384 + k*128 + o). g = mean over K stacks of final out.
// ---------------------------------------------------------------------------
__global__ void __launch_bounds__(256)
arma_iter(const float* __restrict__ rootc, // [2500, 2688]
          const float* __restrict__ armaw, // [5,3,128,128]
          const float* __restrict__ armab, // [6,3,128]
          const int*   __restrict__ aei,   // [2, 98]
          float* __restrict__ g)           // [2500, 128]
{
  __shared__ float sOut[50 * 128];
  __shared__ float sBf[50 * 128];
  __shared__ float sDeg[50];
  __shared__ float sDinv[50];
  __shared__ float sNorm[98];
  int b = blockIdx.x, tid = threadIdx.x;
  if (tid < 50) sDeg[tid] = 0.f;
  __syncthreads();
  if (tid < 98) atomicAdd(&sDeg[aei[98 + tid]], 1.f);
  __syncthreads();
  if (tid < 50) {
    float d = sDeg[tid];
    sDinv[tid] = (d > 0.f) ? rsqrtf(fmaxf(d, 1.f)) : 0.f;
  }
  __syncthreads();
  if (tid < 98) sNorm[tid] = sDinv[aei[tid]] * sDinv[aei[98 + tid]];
  __syncthreads();

  const float kinv = 1.f / 3.f;
  for (int k = 0; k < 3; ++k) {
    for (int idx = tid; idx < 50 * 128; idx += 256) {
      int mm = idx >> 7, o = idx & 127;
      sOut[idx] = rootc[(size_t)(b * 50 + mm) * 2688 + k * 128 + o];
    }
    __syncthreads();
    for (int t = 0; t < 6; ++t) {
      if (t > 0) {
        const float* W = armaw + ((size_t)(t - 1) * 3 + k) * 128 * 128;
        for (int idx = tid; idx < 50 * 128; idx += 256) {
          int mm = idx >> 7, o = idx & 127;
          float s = 0.f;
          const float* row = &sOut[mm * 128];
          for (int p = 0; p < 128; ++p) s += row[p] * W[p * 128 + o]; // coalesced W
          sBf[idx] = s;
        }
      } else {
        for (int idx = tid; idx < 50 * 128; idx += 256) sBf[idx] = sOut[idx];
      }
      __syncthreads();
      for (int idx = tid; idx < 50 * 128; idx += 256) sOut[idx] = 0.f;
      __syncthreads();
      for (int idx = tid; idx < 98 * 128; idx += 256) {
        int e = idx >> 7, o = idx & 127;
        atomicAdd(&sOut[aei[98 + e] * 128 + o], sNorm[e] * sBf[aei[e] * 128 + o]);
      }
      __syncthreads();
      const float* bias = armab + ((size_t)t * 3 + k) * 128;
      for (int idx = tid; idx < 50 * 128; idx += 256) {
        int mm = idx >> 7, o = idx & 127;
        float v = sOut[idx]
                + rootc[(size_t)(b * 50 + mm) * 2688 + 384 + t * 384 + k * 128 + o]
                + bias[o];
        sOut[idx] = fmaxf(v, 0.f);
      }
      __syncthreads();
    }
    for (int idx = tid; idx < 50 * 128; idx += 256) {
      size_t gi = (size_t)b * 50 * 128 + idx;
      if (k == 0) g[gi] = sOut[idx] * kinv;
      else        g[gi] += sOut[idx] * kinv;
    }
    __syncthreads();
  }
}

// ---------------------------------------------------------------------------
// Attention over amino acids + MLP head. One block (128 thr) per graph.
// ---------------------------------------------------------------------------
__global__ void __launch_bounds__(128)
final_mlp(const float* __restrict__ g,   // [2500, 128]
          const float* __restrict__ Wa, const float* __restrict__ ba,
          const float* __restrict__ W1, const float* __restrict__ b1,
          const float* __restrict__ W2, const float* __restrict__ b2,
          const float* __restrict__ W3, const float* __restrict__ b3,
          const float* __restrict__ W4, const float* __restrict__ b4,
          float* __restrict__ out)
{
  __shared__ float sAttn[50];
  __shared__ float sP[128];
  __shared__ float sR1[64];
  __shared__ float sR2[32];
  __shared__ float sR3[16];
  __shared__ float sMx, sSum;
  int b = blockIdx.x, tid = threadIdx.x;
  if (tid < 50) {
    const float* gr = g + (size_t)(b * 50 + tid) * 128;
    float s = ba[0];
    for (int o = 0; o < 128; ++o) s += gr[o] * Wa[o];
    sAttn[tid] = s;
  }
  __syncthreads();
  if (tid == 0) {
    float mx = sAttn[0];
    for (int i = 1; i < 50; ++i) mx = fmaxf(mx, sAttn[i]);
    sMx = mx;
  }
  __syncthreads();
  if (tid < 50) sAttn[tid] = __expf(sAttn[tid] - sMx);
  __syncthreads();
  if (tid == 0) {
    float s = 0.f;
    for (int i = 0; i < 50; ++i) s += sAttn[i];
    sSum = s;
  }
  __syncthreads();
  float inv = 1.f / sSum;
  {
    float acc = 0.f;
    for (int mm = 0; mm < 50; ++mm)
      acc += sAttn[mm] * inv * g[(size_t)(b * 50 + mm) * 128 + tid];
    sP[tid] = acc;
  }
  __syncthreads();
  if (tid < 64) {
    float v = b1[tid];
    for (int f = 0; f < 128; ++f) v += sP[f] * W1[f * 64 + tid];
    sR1[tid] = fmaxf(v, 0.f);
  }
  __syncthreads();
  if (tid < 32) {
    float v = b2[tid];
    for (int f = 0; f < 64; ++f) v += sR1[f] * W2[f * 32 + tid];
    sR2[tid] = fmaxf(v, 0.f);
  }
  __syncthreads();
  if (tid < 16) {
    float v = b3[tid];
    for (int f = 0; f < 32; ++f) v += sR2[f] * W3[f * 16 + tid];
    sR3[tid] = fmaxf(v, 0.f);
  }
  __syncthreads();
  if (tid == 0) {
    float v = b4[0];
    for (int f = 0; f < 16; ++f) v += sR3[f] * W4[f];
    out[b] = v;
  }
}

// ---------------------------------------------------------------------------
extern "C" void kernel_launch(void* const* d_in, const int* in_sizes, int n_in,
                              void* d_out, int out_size, void* d_ws, size_t ws_size,
                              hipStream_t stream)
{
  (void)in_sizes; (void)n_in; (void)out_size; (void)ws_size;
  const float* x     = (const float*)d_in[0];
  const float* ea    = (const float*)d_in[1];
  const float* aaf   = (const float*)d_in[2];
  const int*   eidx  = (const int*)d_in[3];
  const int*   lbl   = (const int*)d_in[4];
  const int*   aei   = (const int*)d_in[5];
  const float* We1   = (const float*)d_in[6];
  const float* be1   = (const float*)d_in[7];
  const float* root1 = (const float*)d_in[8];
  const float* bias1 = (const float*)d_in[9];
  const float* We2   = (const float*)d_in[10];
  const float* be2   = (const float*)d_in[11];
  const float* root2 = (const float*)d_in[12];
  const float* bias2 = (const float*)d_in[13];
  const float* WaA   = (const float*)d_in[14];
  const float* baA   = (const float*)d_in[15];
  const float* initw = (const float*)d_in[16];
  const float* armaw = (const float*)d_in[17];
  const float* rootw = (const float*)d_in[18];
  const float* armab = (const float*)d_in[19];
  const float* Waaa  = (const float*)d_in[20];
  const float* baaa  = (const float*)d_in[21];
  const float* W1 = (const float*)d_in[22]; const float* b1 = (const float*)d_in[23];
  const float* W2 = (const float*)d_in[24]; const float* b2 = (const float*)d_in[25];
  const float* W3 = (const float*)d_in[26]; const float* b3 = (const float*)d_in[27];
  const float* W4 = (const float*)d_in[28]; const float* b4 = (const float*)d_in[29];

  char* ws = (char*)d_ws;
  size_t off = 0;
  auto alloc = [&](size_t bytes) -> void* {
    void* p = ws + off;
    off = (off + bytes + 255) & ~(size_t)255;
    return p;
  };
  float*    agg1 = (float*)alloc((size_t)Ncnt * 32 * 4);
  float*    h1   = (float*)alloc((size_t)Ncnt * 32 * 4);
  float*    agg2 = (float*)alloc((size_t)Ncnt * 64 * 4);
  float*    h2   = (float*)alloc((size_t)Ncnt * 64 * 4);
  float*    h_aa = (float*)alloc((size_t)2500 * 160 * 4);
  float*    rc   = (float*)alloc((size_t)2500 * 2688 * 4);
  _Float16* wbA  = (_Float16*)alloc((size_t)5 * 168 * 32 * 16 * 2);
  float*    gbuf = (float*)alloc((size_t)2500 * 128 * 4);

  zero_f32<<<(Ncnt * 32 + 255) / 256, 256, 0, stream>>>(agg1, Ncnt * 32);
  zero_f32<<<(Ncnt * 64 + 255) / 256, 256, 0, stream>>>(agg2, Ncnt * 64);
  prep_wbarma<<<(5 * 168 * 32 * 16) / 256, 256, 0, stream>>>(initw, rootw, wbA);

  int eblocks = (Ecnt + 127) / 128;
  nnconv_edge_wmma<<<eblocks, 256, 0, stream>>>(x, ea, eidx, We1, be1, agg1, 32, 0);
  node_update<<<(Ncnt + 255) / 256, 256, 0, stream>>>(x, agg1, root1, bias1, h1,
                                                      Ncnt, 32, 32);
  nnconv_edge_wmma<<<eblocks, 256, 0, stream>>>(h1, ea, eidx, We2, be2, agg2, 64, 0);
  nnconv_edge_wmma<<<eblocks, 256, 0, stream>>>(h1, ea, eidx, We2, be2, agg2, 64, 1);
  node_update<<<(Ncnt + 255) / 256, 256, 0, stream>>>(h1, agg2, root2, bias2, h2,
                                                      Ncnt, 32, 64);

  atom_readout<<<50, 256, 0, stream>>>(h2, aaf, lbl, WaA, baA, h_aa);

  dim3 ggrid(157, 11);
  arma_gemm_wmma<<<ggrid, 256, 0, stream>>>(h_aa, wbA, rc);
  arma_iter<<<50, 256, 0, stream>>>(rc, armaw, armab, aei, gbuf);
  final_mlp<<<50, 128, 0, stream>>>(gbuf, Waaa, baaa, W1, b1, W2, b2, W3, b3,
                                    W4, b4, (float*)d_out);
}
